// Model_17746804867311
// MI455X (gfx1250) — matrix operand order repaired and checked by reading, not compiled, not run on hardware
//
#include <hip/hip_runtime.h>
#include <hip/hip_bf16.h>
#include <math.h>

typedef __attribute__((ext_vector_type(16))) _Float16 v16h;
typedef __attribute__((ext_vector_type(8)))  float    v8f;

// ---------------------------------------------------------------------------
// Generic implicit-GEMM conv (and transposed-conv via virtual dilation) using
// V_WMMA_F32_16X16X32_F16.  GEMM view: D[Cout x pixels] = W[Cout x K] * X[K x pixels],
// K = Cin*ks*ks in chunks of 32.  One block = 64-pixel tile (4 rows x 16 px);
// one wave per 16-row Cout tile, 4 WMMAs per wave per K-chunk reusing the
// register-resident A fragment.  Activations are f16 in HBM (flags for the
// fp32 network input / output), accumulation in f32.
// ---------------------------------------------------------------------------
struct ConvP {
    const void*     in;     // f32 or f16 activations (in_f16)
    const _Float16* wpk;    // pre-packed A fragments (f16, WMMA lane layout)
    const float*    bias;   // nullptr if none
    const _Float16* skip;   // nullptr if none (residual add, f16)
    void*           out;    // f16 or f32 (out_f16)
    int Cin, Hin, Win, Cout, Hout, Wout;
    int ks, stride, pad;
    int Ktot, Kchunks;
    int relu_in, relu_out, dilate, in_f16, out_f16;
};

__global__ void conv_wmma(ConvP p) {
    __shared__ alignas(32) _Float16 ldsB[64][32];   // [pixel][k] f16 im2col tile (4 KB)
    const int lane = threadIdx.x & 31;
    const int wv   = threadIdx.x >> 5;              // wave == Cout tile
    const int ox0  = blockIdx.x << 4;
    const int oy0  = blockIdx.y << 2;               // 4 output rows per block
    const int b    = blockIdx.z;
    const int ks2  = p.ks * p.ks;

    v8f acc[4];
#pragma unroll
    for (int t = 0; t < 4; ++t) acc[t] = (v8f){0.f,0.f,0.f,0.f,0.f,0.f,0.f,0.f};

    for (int kc = 0; kc < p.Kchunks; ++kc) {
        __syncthreads();
        // cooperative im2col fill: 2048 halves, uniform loop (2048 % blockDim == 0)
        for (int it = threadIdx.x; it < 2048; it += blockDim.x) {
            const int px = it >> 5;                 // 0..63
            const int k  = it & 31;
            const int py = px >> 4;
            const int oy = oy0 + py;
            const int oxp = ox0 + (px & 15);
            const int kg = (kc << 5) + k;
            float v = 0.f;
            if (kg < p.Ktot) {
                const int cin = kg / ks2;
                const int r   = kg - cin * ks2;
                const int ky  = r / p.ks;
                const int kx  = r - ky * p.ks;
                int iy, ix; bool ok;
                if (p.dilate) { // transposed conv: virtual lhs_dilation=2, pad (1,2)
                    const int ty = oy + ky;
                    const int tx = oxp + kx;
                    iy = (ty - 1) >> 1;
                    ix = (tx - 1) >> 1;
                    ok = (ty & 1) && (tx & 1) && (iy < p.Hin) && (ix < p.Win);
                } else {
                    iy = oy * p.stride + ky - p.pad;
                    ix = oxp * p.stride + kx - p.pad;
                    ok = (iy >= 0) && (iy < p.Hin) && (ix >= 0) && (ix < p.Win);
                }
                if (ok) {
                    const size_t idx = (((size_t)b * p.Cin + cin) * p.Hin + iy) * p.Win + ix;
                    v = p.in_f16 ? (float)((const _Float16*)p.in)[idx]
                                 : ((const float*)p.in)[idx];
                    if (p.relu_in) v = fmaxf(v, 0.f);
                }
            }
            ldsB[px][k] = (_Float16)v;
        }
        __syncthreads();
        // prefetch next K-chunk's A fragment while this chunk computes
        if (kc + 1 < p.Kchunks)
            __builtin_prefetch(p.wpk + (((size_t)wv * p.Kchunks + kc + 1) * 32 + lane) * 16, 0, 1);
        // A fragment: pre-packed per-lane 16 halves (two 16B global loads)
        const v16h a = *(const v16h*)(p.wpk +
                         (((size_t)wv * p.Kchunks + kc) * 32 + lane) * 16);
        // B fragments: lanes 0-15 -> pixel, K 0..15 ; lanes 16-31 -> K 16..31
#pragma unroll
        for (int py = 0; py < 4; ++py) {
            const v16h bb = *(const v16h*)&ldsB[(py << 4) + (lane & 15)][(lane >> 4) << 4];
            acc[py] = __builtin_amdgcn_wmma_f32_16x16x32_f16(false, a, false, bb,
                                                             (short)0, acc[py], false, false);
        }
    }

    // epilogue: C/D layout — VGPR r: lanes0-15 M=r, lanes16-31 M=r+8; N = lane&15
    const int nn = lane & 15;
    const int hi = lane >> 4;
    const int ox = ox0 + nn;
#pragma unroll
    for (int py = 0; py < 4; ++py) {
        const int oy = oy0 + py;
#pragma unroll
        for (int r = 0; r < 8; ++r) {
            const int oc = (wv << 4) + r + (hi << 3);
            if (oc < p.Cout) {
                float v = acc[py][r];
                if (p.bias) v += p.bias[oc];
                const size_t off = (((size_t)b * p.Cout + oc) * p.Hout + oy) * p.Wout + ox;
                if (p.skip) v += (float)p.skip[off];
                if (p.relu_out) v = fmaxf(v, 0.f);
                if (p.out_f16) ((_Float16*)p.out)[off] = (_Float16)v;
                else           ((float*)p.out)[off]    = v;
            }
        }
    }
}

// ---------------------------------------------------------------------------
// Weight packer: fp32 weights -> f16 A fragments in WMMA 16x32 lane layout.
// k index = cin*ks*ks + ky*ks + kx.  convt=1: src is (Cin,Cout,ks,ks), use
// flipped kernel (ConvTranspose2d equivalence).
// ---------------------------------------------------------------------------
__global__ void pack_w(const float* __restrict__ src, _Float16* __restrict__ dst,
                       int Cout, int Cin, int ks, int Kchunks, int convt, int total) {
    const int id = blockIdx.x * blockDim.x + threadIdx.x;
    if (id >= total) return;
    const int frag = id >> 9;
    const int item = id & 511;
    const int lane = item >> 4;
    const int h    = item & 15;
    const int mt   = frag / Kchunks;
    const int kc   = frag - mt * Kchunks;
    const int row  = (mt << 4) + (lane & 15);
    const int hi   = lane >> 4;
    // A 16x32 f16 packed in this code's contiguous k order (matches B): lanes<16: K {0..15}; lanes>=16: K {16..31}
    const int kl   = (hi << 4) + h;
    const int kg   = (kc << 5) + kl;
    const int ks2  = ks * ks;
    const int Ktot = Cin * ks2;
    float v = 0.f;
    if (row < Cout && kg < Ktot) {
        const int cin = kg / ks2;
        const int r   = kg - cin * ks2;
        const int ky  = r / ks;
        const int kx  = r - ky * ks;
        if (convt)
            v = src[(((size_t)cin * Cout + row) * ks + (ks - 1 - ky)) * ks + (ks - 1 - kx)];
        else
            v = src[(((size_t)row * Cin + cin) * ks + ky) * ks + kx];
    }
    dst[id] = (_Float16)v;
}

// ---------------------------------------------------------------------------
// VQ: codebook squared norms
// ---------------------------------------------------------------------------
__global__ void csq_kernel(const float* __restrict__ cb, float* __restrict__ csq) {
    const int i = threadIdx.x;   // 512
    float s = 0.f;
    for (int d = 0; d < 64; ++d) { const float v = cb[i * 64 + d]; s += v * v; }
    csq[i] = s;
}

__global__ void vq_init(float* lossAcc, float* counts) {
    if (threadIdx.x == 0) lossAcc[0] = 0.f;
    counts[threadIdx.x] = 0.f;      // 512 threads
}

// ---------------------------------------------------------------------------
// VQ argmin: per block 16 pixels; 8 waves sweep the 32 codebook tiles with
// WMMA dot products (K=64 -> 2 chunks).  dist = |c|^2 - 2*z.c  (|z|^2 constant).
// z is stored f16 (direct copy into the B tile).
// ---------------------------------------------------------------------------
__global__ void vq_argmin(const _Float16* __restrict__ z, const _Float16* __restrict__ cbpk,
                          const float* __restrict__ csq, int* __restrict__ idxOut) {
    __shared__ alignas(32) _Float16 ldsZ[16][64];
    __shared__ float sminv[8][32];
    __shared__ int   smini[8][32];
    const int lane = threadIdx.x & 31;
    const int wv   = threadIdx.x >> 5;
    const int n0   = blockIdx.x << 4;
    for (int it = threadIdx.x; it < 1024; it += 256) {
        const int px = it >> 6;
        const int d  = it & 63;
        const int n  = n0 + px;
        const int b  = n >> 10;
        const int rm = n & 1023;
        const int y  = rm >> 5, x = rm & 31;
        ldsZ[px][d] = z[(((size_t)b * 64 + d) * 32 + y) * 32 + x];
    }
    __syncthreads();
    float minv = 3.4e38f; int mini = 0;
    const int hi = lane >> 4;
    for (int mt = wv; mt < 32; mt += 8) {
        v8f acc = {0.f,0.f,0.f,0.f,0.f,0.f,0.f,0.f};
#pragma unroll
        for (int kc = 0; kc < 2; ++kc) {
            const v16h a  = *(const v16h*)(cbpk + (((size_t)mt * 2 + kc) * 32 + lane) * 16);
            const v16h bb = *(const v16h*)&ldsZ[lane & 15][(kc << 5) + (hi << 4)];
            acc = __builtin_amdgcn_wmma_f32_16x16x32_f16(false, a, false, bb,
                                                         (short)0, acc, false, false);
        }
#pragma unroll
        for (int r = 0; r < 8; ++r) {
            const int code = (mt << 4) + r + (hi << 3);
            const float d  = csq[code] - 2.f * acc[r];
            if (d < minv) { minv = d; mini = code; }
        }
    }
    sminv[wv][lane] = minv;
    smini[wv][lane] = mini;
    __syncthreads();
    if (threadIdx.x < 16) {   // pixel n covered by lanes n and n+16 of each wave
        float bm = 3.4e38f; int bi = 0x7fffffff;
        for (int w = 0; w < 8; ++w)
            for (int l = (int)threadIdx.x; l < 32; l += 16) {
                const float v = sminv[w][l];
                const int   i = smini[w][l];
                if (v < bm || (v == bm && i < bi)) { bm = v; bi = i; }
            }
        idxOut[n0 + threadIdx.x] = bi;
    }
}

// gather q into NCHW (f16 for the decoder), accumulate SSE and code histogram
__global__ void vq_gather(const _Float16* __restrict__ z, const float* __restrict__ cb,
                          const int* __restrict__ idx, _Float16* __restrict__ qn,
                          float* lossAcc, float* counts) {
    const int n = blockIdx.x * blockDim.x + threadIdx.x;
    if (n >= 32768) return;
    const int b = n >> 10, rm = n & 1023, y = rm >> 5, x = rm & 31;
    const int code = idx[n];
    float s = 0.f;
    for (int d = 0; d < 64; ++d) {
        const float q = cb[code * 64 + d];
        const size_t off = (((size_t)b * 64 + d) * 32 + y) * 32 + x;
        const float df = q - (float)z[off];
        s += df * df;
        qn[off] = (_Float16)q;
    }
    atomicAdd(lossAcc, s);
    atomicAdd(&counts[code], 1.f);
}

__global__ void vq_final(const float* lossAcc, const float* counts,
                         float* out_loss, float* out_perp) {
    __shared__ float sh[512];
    const int i = threadIdx.x;     // 512
    const float p = counts[i] * (1.f / 32768.f);
    sh[i] = p * logf(p + 1e-10f);
    __syncthreads();
    for (int s = 256; s > 0; s >>= 1) {
        if (i < s) sh[i] += sh[i + s];
        __syncthreads();
    }
    if (i == 0) {
        *out_perp = expf(-sh[0]);
        // loss = q_latent + beta*e_latent = 1.25 * mean((q - z)^2)
        *out_loss = 1.25f * lossAcc[0] / (32768.f * 64.f);
    }
}

// ---------------------------------------------------------------------------
extern "C" void kernel_launch(void* const* d_in, const int* in_sizes, int n_in,
                              void* d_out, int out_size, void* d_ws, size_t ws_size,
                              hipStream_t stream) {
    (void)in_sizes; (void)n_in; (void)ws_size;
    float* out      = (float*)d_out;
    float* y_out    = out + 1;
    float* loss_out = out;
    float* perp_out = out + (out_size - 1);

    // ---- inputs (setup_inputs order) ----
    const float* x      = (const float*)d_in[0];
    const float* ec1_b  = (const float*)d_in[2];
    const float* ec2_b  = (const float*)d_in[4];
    const float* ec3_b  = (const float*)d_in[6];
    const float* ec4_b  = (const float*)d_in[8];
    const float* pvq_b  = (const float*)d_in[14];
    const float* cbk    = (const float*)d_in[15];
    const float* dc1_b  = (const float*)d_in[17];
    const float* dt1_b  = (const float*)d_in[23];
    const float* dt2_b  = (const float*)d_in[25];
    const float* dt3_b  = (const float*)d_in[27];

    // ---- workspace carve-up (activations now f16 -> half the bytes) ----
    char* ws = (char*)d_ws;
    size_t off = 0;
    auto take = [&](size_t bytes) -> void* {
        void* p = ws + off;
        off += (bytes + 255) & ~(size_t)255;
        return p;
    };
    _Float16* bufA = (_Float16*)take((size_t)32 * 64 * 128 * 128 * 2);  // 67 MB
    _Float16* bufB = (_Float16*)take((size_t)32 * 128 * 64 * 64 * 2);   // 33 MB
    _Float16* bufC = (_Float16*)take((size_t)32 * 128 * 32 * 32 * 2);
    _Float16* bufD = (_Float16*)take((size_t)32 * 128 * 32 * 32 * 2);
    _Float16* bufT = (_Float16*)take((size_t)32 * 32 * 32 * 32 * 2);
    _Float16* zbuf = (_Float16*)take((size_t)32 * 64 * 32 * 32 * 2);
    _Float16* qbuf = (_Float16*)take((size_t)32 * 64 * 32 * 32 * 2);
    int*      idxb = (int*)take(32768 * 4);
    float*    csqb = (float*)take(512 * 4);
    float*    cnts = (float*)take(512 * 4);
    float*    lacc = (float*)take(256);

    auto cdiv = [](size_t a, size_t b) { return (a + b - 1) / b; };
    auto pkalloc = [&](int Cout, int Cin, int ks) -> _Float16* {
        size_t halves = cdiv(Cout, 16) * cdiv((size_t)Cin * ks * ks, 32) * 512;
        return (_Float16*)take(halves * 2);
    };
    auto packW = [&](const void* src, _Float16* dst, int Cout, int Cin, int ks, int convt) {
        const int Kc = (int)cdiv((size_t)Cin * ks * ks, 32);
        const int Mt = (int)cdiv(Cout, 16);
        const int total = Mt * Kc * 512;
        pack_w<<<dim3((unsigned)cdiv(total, 256)), dim3(256), 0, stream>>>(
            (const float*)src, dst, Cout, Cin, ks, Kc, convt, total);
    };

    _Float16* pk_ec1  = pkalloc(64, 3, 3);    packW(d_in[1],  pk_ec1, 64, 3, 3, 0);
    _Float16* pk_ec2  = pkalloc(128, 64, 3);  packW(d_in[3],  pk_ec2, 128, 64, 3, 0);
    _Float16* pk_ec3  = pkalloc(128, 128, 3); packW(d_in[5],  pk_ec3, 128, 128, 3, 0);
    _Float16* pk_ec4  = pkalloc(128, 128, 3); packW(d_in[7],  pk_ec4, 128, 128, 3, 0);
    _Float16* pk_er0a = pkalloc(32, 128, 3);  packW(d_in[9],  pk_er0a, 32, 128, 3, 0);
    _Float16* pk_er0b = pkalloc(128, 32, 1);  packW(d_in[10], pk_er0b, 128, 32, 1, 0);
    _Float16* pk_er1a = pkalloc(32, 128, 3);  packW(d_in[11], pk_er1a, 32, 128, 3, 0);
    _Float16* pk_er1b = pkalloc(128, 32, 1);  packW(d_in[12], pk_er1b, 128, 32, 1, 0);
    _Float16* pk_pvq  = pkalloc(64, 128, 1);  packW(d_in[13], pk_pvq, 64, 128, 1, 0);
    _Float16* pk_cb   = pkalloc(512, 64, 1);  packW(d_in[15], pk_cb, 512, 64, 1, 0);
    _Float16* pk_dc1  = pkalloc(128, 64, 3);  packW(d_in[16], pk_dc1, 128, 64, 3, 0);
    _Float16* pk_dr0a = pkalloc(32, 128, 3);  packW(d_in[18], pk_dr0a, 32, 128, 3, 0);
    _Float16* pk_dr0b = pkalloc(128, 32, 1);  packW(d_in[19], pk_dr0b, 128, 32, 1, 0);
    _Float16* pk_dr1a = pkalloc(32, 128, 3);  packW(d_in[20], pk_dr1a, 32, 128, 3, 0);
    _Float16* pk_dr1b = pkalloc(128, 32, 1);  packW(d_in[21], pk_dr1b, 128, 32, 1, 0);
    _Float16* pk_dt1  = pkalloc(64, 128, 3);  packW(d_in[22], pk_dt1, 64, 128, 3, 1);
    _Float16* pk_dt2  = pkalloc(64, 64, 3);   packW(d_in[24], pk_dt2, 64, 64, 3, 1);
    _Float16* pk_dt3  = pkalloc(3, 64, 3);    packW(d_in[26], pk_dt3, 3, 64, 3, 1);

    csq_kernel<<<dim3(1), dim3(512), 0, stream>>>(cbk, csqb);
    vq_init<<<dim3(1), dim3(512), 0, stream>>>(lacc, cnts);

    auto conv = [&](const void* in, const _Float16* wpk, const float* bias,
                    const _Float16* skip, void* o,
                    int Cin, int Hin, int Win, int Cout, int Hout, int Wout,
                    int ks, int stride, int pad,
                    int relu_in, int relu_out, int dilate, int in_f16, int out_f16) {
        ConvP p;
        p.in = in; p.wpk = wpk; p.bias = bias; p.skip = skip; p.out = o;
        p.Cin = Cin; p.Hin = Hin; p.Win = Win;
        p.Cout = Cout; p.Hout = Hout; p.Wout = Wout;
        p.ks = ks; p.stride = stride; p.pad = pad;
        p.Ktot = Cin * ks * ks;
        p.Kchunks = (p.Ktot + 31) / 32;
        p.relu_in = relu_in; p.relu_out = relu_out; p.dilate = dilate;
        p.in_f16 = in_f16; p.out_f16 = out_f16;
        const int Mt = (Cout + 15) / 16;
        dim3 grid((unsigned)(Wout / 16), (unsigned)(Hout / 4), 32u);
        conv_wmma<<<grid, dim3((unsigned)(32 * Mt)), 0, stream>>>(p);
    };

    // ---- Encoder ----
    conv(x,    pk_ec1, ec1_b, nullptr, bufA,   3, 256, 256,  64, 128, 128, 3, 2, 1, 0, 1, 0, 0, 1);
    conv(bufA, pk_ec2, ec2_b, nullptr, bufB,  64, 128, 128, 128,  64,  64, 3, 2, 1, 0, 1, 0, 1, 1);
    conv(bufB, pk_ec3, ec3_b, nullptr, bufC, 128,  64,  64, 128,  32,  32, 3, 2, 1, 0, 1, 0, 1, 1);
    conv(bufC, pk_ec4, ec4_b, nullptr, bufD, 128,  32,  32, 128,  32,  32, 3, 1, 1, 0, 0, 0, 1, 1);
    // residual stack (relu fused into consumer's relu_in)
    conv(bufD, pk_er0a, nullptr, nullptr, bufT, 128, 32, 32,  32, 32, 32, 3, 1, 1, 1, 0, 0, 1, 1);
    conv(bufT, pk_er0b, nullptr, bufD,    bufC,  32, 32, 32, 128, 32, 32, 1, 1, 0, 1, 0, 0, 1, 1);
    conv(bufC, pk_er1a, nullptr, nullptr, bufT, 128, 32, 32,  32, 32, 32, 3, 1, 1, 1, 0, 0, 1, 1);
    conv(bufT, pk_er1b, nullptr, bufC,    bufD,  32, 32, 32, 128, 32, 32, 1, 1, 0, 1, 0, 0, 1, 1);
    // pre-VQ 1x1 (final stack relu fused as relu_in)
    conv(bufD, pk_pvq, pvq_b, nullptr, zbuf, 128, 32, 32, 64, 32, 32, 1, 1, 0, 1, 0, 0, 1, 1);

    // ---- Vector quantization ----
    vq_argmin<<<dim3(2048), dim3(256), 0, stream>>>(zbuf, pk_cb, csqb, idxb);
    vq_gather<<<dim3(128), dim3(256), 0, stream>>>(zbuf, cbk, idxb, qbuf, lacc, cnts);
    vq_final<<<dim3(1), dim3(512), 0, stream>>>(lacc, cnts, loss_out, perp_out);

    // ---- Decoder ----
    conv(qbuf, pk_dc1, dc1_b, nullptr, bufC, 64, 32, 32, 128, 32, 32, 3, 1, 1, 0, 0, 0, 1, 1);
    conv(bufC, pk_dr0a, nullptr, nullptr, bufT, 128, 32, 32,  32, 32, 32, 3, 1, 1, 1, 0, 0, 1, 1);
    conv(bufT, pk_dr0b, nullptr, bufC,    bufD,  32, 32, 32, 128, 32, 32, 1, 1, 0, 1, 0, 0, 1, 1);
    conv(bufD, pk_dr1a, nullptr, nullptr, bufT, 128, 32, 32,  32, 32, 32, 3, 1, 1, 1, 0, 0, 1, 1);
    conv(bufT, pk_dr1b, nullptr, bufD,    bufC,  32, 32, 32, 128, 32, 32, 1, 1, 0, 1, 0, 0, 1, 1);
    // transposed convs via virtual dilation (res-stack relu fused into dt1 relu_in)
    conv(bufC, pk_dt1, dt1_b, nullptr, bufB, 128, 32, 32, 64,  64,  64, 3, 1, 1, 1, 1, 1, 1, 1);
    conv(bufB, pk_dt2, dt2_b, nullptr, bufA,  64, 64, 64, 64, 128, 128, 3, 1, 1, 0, 1, 1, 1, 1);
    conv(bufA, pk_dt3, dt3_b, nullptr, y_out, 64, 128, 128, 3, 256, 256, 3, 1, 1, 0, 0, 1, 1, 0);
}
